// EncoderText_8340826488886
// MI455X (gfx1250) — matrix operand order repaired and checked by reading, not compiled, not run on hardware
//
#include <hip/hip_runtime.h>
#include <hip/hip_bf16.h>
#include <math.h>
#include <stdint.h>

typedef __attribute__((ext_vector_type(16))) _Float16 v16h;
typedef __attribute__((ext_vector_type(8)))  float    v8f;

#define Bn   256
#define Sn   64
#define Dn   512
#define HIDn 128
#define NEGC 1e10f
#define VOCABn 50000

// d_out flat offsets (floats), reference tuple order
#define OFF_FEATS   0
#define OFF_LEFTF   8257536      // 63*256*512
#define OFF_RIGHTF  16515072
#define OFF_OUTWORD 24772608
#define OFF_IDX     33161216     // + 256*64*512
#define OFF_PROB    33177344
#define OFF_SPAN    33193472

#define SMEM_BYTES  302752
#define SYN_LDS_OFF 0u               // syn_s byte offset in dynamic LDS
#define SEM_LDS_OFF (Sn * Dn * 4u)   // sem_s byte offset

#if defined(__HIP_DEVICE_COMPILE__) && __has_builtin(__builtin_amdgcn_tensor_load_to_lds) && __has_builtin(__builtin_amdgcn_s_wait_tensorcnt)
#define HAVE_TDM 1
#else
#define HAVE_TDM 0
#endif

#if HAVE_TDM
typedef unsigned int v4u __attribute__((ext_vector_type(4)));
typedef int          v4i __attribute__((ext_vector_type(4)));
typedef int          v8i __attribute__((ext_vector_type(8)));

// Gather 16 embedding rows (f32, 512 wide) into LDS via the Tensor Data Mover.
// D# built per CDNA5 ISA 8.3-8.7: gather_mode=1, 16-bit indices, data_size=4B.
__device__ __forceinline__ void tdm_gather16(const float* gbase, unsigned lds_byte_off,
                                             const int* toks, int first) {
    unsigned long long ga = (unsigned long long)(uintptr_t)gbase;
    v4u g0;
    g0[0] = 1u | (0u << 30) | (1u << 31);                       // count=1, idx16, gather_mode
    g0[1] = lds_byte_off;                                       // lds_addr
    g0[2] = (unsigned)(ga & 0xffffffffu);                       // global_addr[31:0]
    g0[3] = (unsigned)((ga >> 32) & 0x01ffffffu) | (2u << 30);  // addr[56:32], type=2
    v8i g1;
    g1[0] = (int)(2u << 16);                                    // data_size=4B, wg_mask=0
    g1[1] = (int)((Dn & 0xffffu) << 16);                        // tensor_dim0[15:0]=512
    g1[2] = (int)((VOCABn & 0xffffu) << 16);                    // dim0 hi=0, tensor_dim1 lo=50000
    g1[3] = (int)((unsigned)Dn << 16);                          // dim1 hi=0, tile_dim0=512
    g1[4] = 16;                                                 // tile_dim1 = #indices
    g1[5] = Dn;                                                 // tensor_dim0_stride = 512
    g1[6] = 0;
    g1[7] = 0;
    v4i g2, g3;
    #pragma unroll
    for (int w = 0; w < 4; ++w) {
        int i0 = __builtin_amdgcn_readfirstlane(toks[first + 2 * w]);
        int i1 = __builtin_amdgcn_readfirstlane(toks[first + 2 * w + 1]);
        int j0 = __builtin_amdgcn_readfirstlane(toks[first + 8 + 2 * w]);
        int j1 = __builtin_amdgcn_readfirstlane(toks[first + 8 + 2 * w + 1]);
        g2[w] = (i0 & 0xffff) | (i1 << 16);
        g3[w] = (j0 & 0xffff) | (j1 << 16);
    }
#if __clang_major__ >= 23
    v8i g4 = {0, 0, 0, 0, 0, 0, 0, 0};
    __builtin_amdgcn_tensor_load_to_lds(g0, g1, g2, g3, g4, 0);
#else
    __builtin_amdgcn_tensor_load_to_lds(g0, g1, g2, g3, 0);
#endif
}
#endif

// ---------------- W1 f32 -> f16 (row-major, 128x512) ----------------
__global__ void w1_half_kernel(const float* __restrict__ W1, _Float16* __restrict__ W1h) {
    int i = blockIdx.x * 256 + threadIdx.x;   // 65536 total
    W1h[i] = (_Float16)W1[i];
}

// ---------------- embedding gather + masked out_word ----------------
__global__ void embed_kernel(const int* __restrict__ x, const int* __restrict__ lengths,
                             const float* __restrict__ emb, float* __restrict__ out_word) {
    int bs = blockIdx.x;                  // 0..B*S-1
    int b = bs >> 6, s = bs & 63;
    int t = threadIdx.x;                  // 128 threads, one float4 each
    float m = (s < lengths[b]) ? 1.0f : 0.0f;
    float4 v = ((const float4*)emb)[(size_t)x[bs] * (Dn / 4) + t];
    ((float4*)(out_word + (size_t)bs * Dn))[t] = make_float4(v.x * m, v.y * m, v.z * m, v.w * m);
}

// ---------------- resident greedy parser: 1 workgroup per batch ----------------
__global__ __launch_bounds__(256)
void parse_kernel(const int* __restrict__ x, const int* __restrict__ lengths,
                  const float* __restrict__ emb, const _Float16* __restrict__ W1h,
                  const float* __restrict__ b1, const float* __restrict__ W2,
                  const float* __restrict__ scoreW, float* __restrict__ out) {
    extern __shared__ float smem[];
    float* syn_s  = smem;                  // 64*512   (byte off 0)
    float* sem_s  = syn_s  + Sn * Dn;      // 64*512   (byte off 131072)
    float* hbuf   = sem_s  + Sn * Dn;      // 64*128
    float* red    = hbuf   + Sn * HIDn;    // 256*4
    float* tag_s  = red    + 1024;         // 64*2
    float* logit  = tag_s  + 128;          // 64
    float* rnorm  = logit  + 64;           // 64
    float* lb     = rnorm  + 64;           // 64
    float* rb     = lb     + 64;           // 64
    float* b1s    = rb     + 64;           // 128
    float* w2s    = b1s    + 128;          // 256
    float* sws    = w2s    + 256;          // 8 (4 used)
    float* nrm    = sws    + 8;            // 8 (4 used)
    float* fmisc  = nrm    + 8;            // 8
    int*   perm   = (int*)(fmisc + 8);     // 64
    int*   imisc  = perm   + 64;           // 16
    int*   toks   = imisc  + 16;           // 64

    const int tid  = threadIdx.x;
    const int b    = blockIdx.x;
    const int lane = tid & 31;
    const int wave = tid >> 5;
    const int mylen = lengths[b];

    // Warm L2 with W1h (shared by all 256 blocks): global_prefetch_b8
    __builtin_prefetch((const char*)W1h + tid * 512, 0, 3);
    __builtin_prefetch((const char*)W1h + tid * 512 + 256, 0, 3);

    if (tid < 128) b1s[tid] = b1[tid];
    if (tid < 256) w2s[tid] = W2[tid];
    if (tid < 4)   sws[tid] = scoreW[tid];
    if (tid < 64) { perm[tid] = tid; lb[tid] = (float)tid; rb[tid] = (float)tid; toks[tid] = x[b * Sn + tid]; }
    __syncthreads();

#if HAVE_TDM
    // TDM gather-mode: 8 tensor ops move 64 scattered 2KB embedding rows into LDS.
    if (wave == 0) {
        #pragma unroll
        for (int g = 0; g < 4; ++g) {
            tdm_gather16(emb, SYN_LDS_OFF + (unsigned)g * 16u * (Dn * 4u), toks, g * 16);
            tdm_gather16(emb, SEM_LDS_OFF + (unsigned)g * 16u * (Dn * 4u), toks, g * 16);
        }
        __builtin_amdgcn_s_wait_tensorcnt(0);
    }
    __syncthreads();
#else
    // Fallback: per-lane float4 gather, 64 rows x 128 float4
    for (int q = tid; q < Sn * (Dn / 4); q += 256) {
        int r = q >> 7, c = q & 127;
        float4 v4 = ((const float4*)emb)[(size_t)toks[r] * (Dn / 4) + c];
        ((float4*)(sem_s + r * Dn))[c] = v4;
        ((float4*)(syn_s + r * Dn))[c] = v4;
    }
    __syncthreads();
#endif

    for (int it = 0; it < Sn - 1; ++it) {
        const int L  = Sn - it;
        const int vc = max(mylen - 1 - it, 0);   // count of valid merge positions

        // ---- 1) inverse L2 norms of logical rows 0..L-1 (4 threads per row)
        {
            int r = tid >> 2, sub = tid & 3;
            float s = 0.0f;
            if (r < L) {
                const float* p = syn_s + perm[r] * Dn + sub;
                #pragma unroll 4
                for (int j = 0; j < Dn / 4; ++j) { float v = p[4 * j]; s += v * v; }
            }
            red[tid] = s;
        }
        __syncthreads();
        if (tid < L) rnorm[tid] = rsqrtf(red[tid * 4] + red[tid * 4 + 1] + red[tid * 4 + 2] + red[tid * 4 + 3]);
        __syncthreads();

        // ---- 2) tagger stage 1 via WMMA: hbuf[m][n] = relu((norm_syn @ W1^T)[m][n] + b1[n])
        {
            const int Mtiles = (L + 15) >> 4;
            const int tiles  = Mtiles * (HIDn / 16);
            const int kh  = lane >> 4;   // lane half
            const int l15 = lane & 15;
            for (int t = wave; t < tiles; t += 8) {
                const int mtile = t >> 3;
                const int ntile = t & 7;
                const int mrow  = mtile * 16 + l15;
                const bool mv   = (mrow < L);
                const float rn  = mv ? rnorm[mrow] : 0.0f;          // zero-pads tail rows
                const float* arow = syn_s + (mv ? perm[mrow] : 0) * Dn;
                const _Float16* brow = W1h + (size_t)(ntile * 16 + l15) * Dn + kh * 16;
                v8f acc = {0.f, 0.f, 0.f, 0.f, 0.f, 0.f, 0.f, 0.f};
                for (int kb = 0; kb < Dn; kb += 32) {
                    // A fragment: contiguous K (lanes 0-15: K 0-15; 16-31: K 16-31), matching B
                    v16h a;
                    #pragma unroll
                    for (int j = 0; j < 8; ++j) {
                        int kk = kb + kh * 16 + 2 * j;
                        a[2 * j]     = (_Float16)(arow[kk]     * rn);
                        a[2 * j + 1] = (_Float16)(arow[kk + 1] * rn);
                    }
                    // B fragment: lanes 0-15 hold K 0..15, lanes 16-31 hold K 16..31, col = l15
                    v16h bf = *(const v16h*)(brow + kb);
                    acc = __builtin_amdgcn_wmma_f32_16x16x32_f16(false, a, false, bf,
                                                                 (short)0, acc, false, false);
                }
                // C layout: vgpr v -> m_local = v + 8*half, n_local = l15.
                // Branchless store: m < Mtiles*16 <= 64 always; rows >= L hold junk but
                // are never read (stage 3 only touches m < L).
                const int n = ntile * 16 + l15;
                const float bias = b1s[n];
                #pragma unroll
                for (int v = 0; v < 8; ++v) {
                    int m = mtile * 16 + 8 * kh + v;
                    hbuf[m * HIDn + n] = fmaxf(acc[v] + bias, 0.0f);
                }
            }
        }
        __syncthreads();

        // ---- 3) tagger stage 2: tag[m][j] = hbuf[m] . W2[j]   (128x2 tiny GEMV)
        if (tid < 2 * Sn) {
            int m = tid >> 1, j = tid & 1;
            if (m < L) {
                const float* h = hbuf + m * HIDn;
                const float* w = w2s + j * HIDn;
                float s = 0.0f;
                #pragma unroll 8
                for (int k = 0; k < HIDn; ++k) s += h[k] * w[k];
                tag_s[tid] = s;
            }
        }
        __syncthreads();

        // ---- 4) pair logits over positions 0..L-2
        if (tid < L - 1) {
            float lg = sws[0] * tag_s[tid * 2]       + sws[1] * tag_s[tid * 2 + 1]
                     + sws[2] * tag_s[(tid + 1) * 2] + sws[3] * tag_s[(tid + 1) * 2 + 1];
            if (tid >= vc) lg -= NEGC;
            logit[tid] = lg;
        }
        __syncthreads();

        // ---- 5) softmax / first-argmax + scalar outputs
        if (tid == 0) {
            int n = L - 1;
            float mx = logit[0]; int am = 0;
            for (int p = 1; p < n; ++p) if (logit[p] > mx) { mx = logit[p]; am = p; }
            float sum = 0.0f;
            for (int p = 0; p < n; ++p) sum += __expf(logit[p] - mx);
            out[OFF_IDX  + (size_t)it * Bn + b] = (float)am;
            out[OFF_PROB + (size_t)it * Bn + b] = 1.0f / sum;           // exp(0)/sum
            out[OFF_SPAN + ((size_t)it * Bn + b) * 2 + 0] = lb[am];
            out[OFF_SPAN + ((size_t)it * Bn + b) * 2 + 1] = rb[am + 1];
            imisc[0] = am;
            fmisc[0] = (vc > 0) ? 1.0f : 0.0f;                          // undone
        }
        __syncthreads();

        const int am = imisc[0];
        const float undone = fmisc[0];
        const int pa = perm[am];
        const int pb = perm[am + 1];

        // ---- 6) norms of sem_a, sem_b, sem_a+sem_b, syn_a+syn_b
        {
            float sa = 0, sb = 0, sc = 0, sy = 0;
            #pragma unroll
            for (int e = 0; e < 2; ++e) {
                int k = tid + 256 * e;
                float xa = sem_s[pa * Dn + k], xb = sem_s[pb * Dn + k];
                float ya = syn_s[pa * Dn + k], yb = syn_s[pb * Dn + k];
                sa += xa * xa; sb += xb * xb;
                float xc = xa + xb; sc += xc * xc;
                float yc = ya + yb; sy += yc * yc;
            }
            red[tid * 4 + 0] = sa; red[tid * 4 + 1] = sb;
            red[tid * 4 + 2] = sc; red[tid * 4 + 3] = sy;
        }
        for (int s = 128; s >= 1; s >>= 1) {
            __syncthreads();
            if (tid < s) {
                red[tid * 4 + 0] += red[(tid + s) * 4 + 0];
                red[tid * 4 + 1] += red[(tid + s) * 4 + 1];
                red[tid * 4 + 2] += red[(tid + s) * 4 + 2];
                red[tid * 4 + 3] += red[(tid + s) * 4 + 3];
            }
        }
        __syncthreads();
        if (tid == 0) {
            nrm[0] = rsqrtf(red[0]); nrm[1] = rsqrtf(red[1]);
            nrm[2] = rsqrtf(red[2]); nrm[3] = rsqrtf(red[3]);
        }
        __syncthreads();

        // ---- 7) feature outputs + in-place merge into slot pa
        {
            const float ia = nrm[0], ib = nrm[1], ic = nrm[2], iy = nrm[3];
            const size_t o = ((size_t)it * Bn + b) * Dn;
            #pragma unroll
            for (int e = 0; e < 2; ++e) {
                int k = tid + 256 * e;
                float xa = sem_s[pa * Dn + k], xb = sem_s[pb * Dn + k];
                float ya = syn_s[pa * Dn + k], yb = syn_s[pb * Dn + k];
                float f  = (xa + xb) * ic;     // l2norm(sem_a+sem_b); l2norm(f)==f
                float fy = (ya + yb) * iy;
                out[OFF_FEATS  + o + k] = f * undone;
                out[OFF_LEFTF  + o + k] = xa * ia * undone;
                out[OFF_RIGHTF + o + k] = xb * ib * undone;
                sem_s[pa * Dn + k] = f;
                syn_s[pa * Dn + k] = fy;
            }
        }
        __syncthreads();

        // ---- 8) logical shift of permutation + bounds (O(64), no row memmove)
        if (tid == 0) {
            rb[am] = rb[am + 1];
            for (int p = am + 1; p < L - 1; ++p) {
                perm[p] = perm[p + 1]; lb[p] = lb[p + 1]; rb[p] = rb[p + 1];
            }
        }
        __syncthreads();
    }
}

extern "C" void kernel_launch(void* const* d_in, const int* in_sizes, int n_in,
                              void* d_out, int out_size, void* d_ws, size_t ws_size,
                              hipStream_t stream) {
    (void)in_sizes; (void)n_in; (void)out_size; (void)ws_size;
    const int*   x    = (const int*)d_in[0];
    const int*   len  = (const int*)d_in[1];
    const float* emb  = (const float*)d_in[2];
    const float* W1   = (const float*)d_in[3];
    const float* b1   = (const float*)d_in[4];
    const float* W2   = (const float*)d_in[5];
    const float* sw   = (const float*)d_in[6];
    float*       out  = (float*)d_out;
    _Float16*    W1h  = (_Float16*)d_ws;     // 128*512 f16 = 128 KB scratch

    w1_half_kernel<<<256, 256, 0, stream>>>(W1, W1h);
    embed_kernel<<<Bn * Sn, 128, 0, stream>>>(x, len, emb, out + OFF_OUTWORD);
    parse_kernel<<<Bn, 256, SMEM_BYTES, stream>>>(x, len, emb, W1h, b1, W2, sw, out);
}
